// Multihead_self_attention_69114613727773
// MI455X (gfx1250) — compile-verified, matrix operand order checked by reading, not run on hardware
//
#include <hip/hip_runtime.h>
#include <hip/hip_bf16.h>

typedef unsigned short u16;
typedef __attribute__((ext_vector_type(16))) __bf16 v16bf;
typedef __attribute__((ext_vector_type(8)))  float  v8f;
typedef __attribute__((ext_vector_type(4)))  unsigned int v4u;
typedef __attribute__((ext_vector_type(8)))  int v8i;
typedef __attribute__((ext_vector_type(4)))  int v4i;

__device__ __forceinline__ u16 f2bf(float f) {
  unsigned int b = __float_as_uint(f);
  b += 0x7FFFu + ((b >> 16) & 1u);
  return (u16)(b >> 16);
}

union FragU { uint4 q[2]; v16bf v; };

// Load a 16-element bf16 WMMA fragment as two contiguous 16B chunks.
// CDNA5 16-bit A/B layout: lane<16 holds K=[0..7],[16..23]; lane>=16 holds
// K=[8..15],[24..31] -> chunk base = (lane>>4)*8, second chunk +16 elements.
__device__ __forceinline__ v16bf load_frag(const u16* p0, const u16* p1) {
  FragU f;
  f.q[0] = *reinterpret_cast<const uint4*>(p0);
  f.q[1] = *reinterpret_cast<const uint4*>(p1);
  return f.v;
}

// TDM: issue a 2D bf16 tile DMA (rows x 64 elts, global row stride = strideElts)
// into LDS at ldsByteAddr with +16-element padding per row (LDS stride 80).
// D# packing per CDNA5 ISA.
__device__ __forceinline__ void tdm_load_tile_k64(
    const u16* gsrc, unsigned ldsByteAddr, int strideElts) {
  unsigned long long ga = (unsigned long long)gsrc;
  v4u g0 = {
      1u,                                   // count=1, user descriptor
      ldsByteAddr,                          // lds_addr (bytes)
      (unsigned)ga,                         // global_addr[31:0]
      (unsigned)((ga >> 32) & 0x1FFFFFFull) | 0x80000000u  // addr[56:32] | type=2
  };
  // d0: data_size=1(2B)<<16 | pad_enable<<20 | pad_interval=4(32DW)<<22
  //     | pad_amount=7(8DW)<<25
  // tensor_dim0=1024, tensor_dim1=1<<20 (no clipping), tile=64x64,
  // tensor_dim0_stride=strideElts.
  v8i g1 = {
      (int)0x0F110000u,
      (int)0x04000000u,                     // tensor_dim0[15:0]=1024 in [31:16]
      0,                                    // tensor_dim0 hi | tensor_dim1 lo
      (int)((1u << 4) | (64u << 16)),       // tensor_dim1 hi=16 | tile_dim0=64
      64,                                   // tile_dim1=64 | tile_dim2=0
      strideElts,                           // tensor_dim0_stride[31:0]
      0, 0
  };
  v4i z4 = {0, 0, 0, 0};
#if __has_include(<hip/amd_detail/amd_gfx1250_TDM.h>)
  v8i z8 = {0, 0, 0, 0, 0, 0, 0, 0};
  __builtin_amdgcn_tensor_load_to_lds(g0, g1, z4, z4, z8, 0);   // 6-arg toolchain
#else
  __builtin_amdgcn_tensor_load_to_lds(g0, g1, z4, z4, 0);       // 5-arg ROCm 7.2
#endif
}

// ---------------------------------------------------------------- fp32->bf16
__global__ void cvt_f32_bf16(const float* __restrict__ in, u16* __restrict__ out, int n4) {
  int i = blockIdx.x * blockDim.x + threadIdx.x;
  int stride = gridDim.x * blockDim.x;
  const float4* in4 = reinterpret_cast<const float4*>(in);
  ushort4* out4 = reinterpret_cast<ushort4*>(out);
  for (; i < n4; i += stride) {
    float4 v = in4[i];
    ushort4 o;
    o.x = f2bf(v.x); o.y = f2bf(v.y); o.z = f2bf(v.z); o.w = f2bf(v.w);
    out4[i] = o;
  }
}

// ------------------------------------------------------- Y = X * W^T (WMMA)
// Block = 128 thr = 4 waves (2x2); wave tile 64x64 (4x4 WMMA tiles) ->
// 16 WMMAs per 8 fragment loads per K-step of 32. Block tile 128x128.
template <int OUT_F32>
__global__ __launch_bounds__(128) void gemm_xwT(
    const u16* __restrict__ A, const u16* __restrict__ W,
    void* __restrict__ Cout, int M, int N, int Kd, float oscale) {
  const int lane = threadIdx.x & 31;
  const int wave = threadIdx.x >> 5;
  const int hl   = lane >> 4;
  const int l15  = lane & 15;
  const int m0 = blockIdx.y * 128 + (wave >> 1) * 64;
  const int n0 = blockIdx.x * 128 + (wave & 1) * 64;

  const v8f vzero = {};
  v8f acc[4][4];
#pragma unroll
  for (int mt = 0; mt < 4; ++mt)
#pragma unroll
    for (int nt = 0; nt < 4; ++nt) acc[mt][nt] = vzero;

  const u16* ap[4];
  const u16* bp[4];
#pragma unroll
  for (int mt = 0; mt < 4; ++mt)
    ap[mt] = A + (size_t)(m0 + mt * 16 + l15) * Kd + hl * 8;
#pragma unroll
  for (int nt = 0; nt < 4; ++nt)
    bp[nt] = W + (size_t)(n0 + nt * 16 + l15) * Kd + hl * 8;

  for (int k0 = 0; k0 < Kd; k0 += 32) {
    v16bf af[4], bfr[4];
#pragma unroll
    for (int mt = 0; mt < 4; ++mt) af[mt] = load_frag(ap[mt] + k0, ap[mt] + k0 + 16);
#pragma unroll
    for (int nt = 0; nt < 4; ++nt) bfr[nt] = load_frag(bp[nt] + k0, bp[nt] + k0 + 16);
    __builtin_prefetch(ap[0] + k0 + 128, 0, 3);   // global_prefetch_b8
    __builtin_prefetch(bp[0] + k0 + 128, 0, 3);
#pragma unroll
    for (int mt = 0; mt < 4; ++mt)
#pragma unroll
      for (int nt = 0; nt < 4; ++nt)
        acc[mt][nt] = __builtin_amdgcn_wmma_f32_16x16x32_bf16(
            false, af[mt], false, bfr[nt], (short)0, acc[mt][nt], false, false);
  }

  // C/D layout: VGPR r -> row r + 8*(lane>>4); col = lane&15
#pragma unroll
  for (int mt = 0; mt < 4; ++mt)
#pragma unroll
    for (int nt = 0; nt < 4; ++nt)
#pragma unroll
      for (int r = 0; r < 8; ++r) {
        int row = m0 + mt * 16 + r + 8 * hl;
        int col = n0 + nt * 16 + l15;
        float v = acc[mt][nt][r] * oscale;
        if (OUT_F32) ((float*)Cout)[(size_t)row * N + col] = v;
        else         ((u16*)Cout)[(size_t)row * N + col] = f2bf(v);
      }
}

// ---------------------------------------------------- causal flash attention
// Grid: (B*H, S/128). Block: 256 thr = 8 waves; wave w owns q-rows [q0, q0+16).
// K tile DMA'd into LDS by the Tensor Data Mover (row pad -> LDS stride 80);
// V staged transposed by the block so P@V B-fragments are contiguous.
// Q is pre-scaled by 1/sqrt(dk) in the Q-projection epilogue.
__global__ __launch_bounds__(256) void flash_attn(
    const u16* __restrict__ Qb, const u16* __restrict__ Kp,
    const u16* __restrict__ Vp, u16* __restrict__ Ob) {
  constexpr int S = 2048, D = 1024, DK = 64, LP = 80;
  const int lane = threadIdx.x & 31;
  const int wave = threadIdx.x >> 5;
  const int hl = lane >> 4;
  const int l15 = lane & 15;
  const int b = blockIdx.x >> 4;
  const int h = blockIdx.x & 15;
  const int qbase = blockIdx.y * 128;
  const int q0 = qbase + wave * 16;

  __shared__ u16 kt[64 * LP];        // K tile, [key][d]  (TDM destination)
  __shared__ u16 vt[64 * LP];        // V tile transposed, [d][key]
  __shared__ u16 pt[8 * 16 * LP];    // per-wave P tiles, [qrow][key]

  const size_t bhoff = (size_t)b * S * D + (size_t)h * DK;

  // Q fragment is invariant across k-tiles: load once (2 K-steps of 32)
  v16bf qf[2];
#pragma unroll
  for (int ks = 0; ks < 2; ++ks) {
    const u16* qp = Qb + bhoff + (size_t)(q0 + l15) * D + ks * 32 + hl * 8;
    qf[ks] = load_frag(qp, qp + 16);
  }

  const v8f vzero = {};
  float mrow[8], lrow[8];
  v8f oacc[4];
#pragma unroll
  for (int r = 0; r < 8; ++r) { mrow[r] = -3.0e38f; lrow[r] = 0.0f; }
#pragma unroll
  for (int nt = 0; nt < 4; ++nt) oacc[nt] = vzero;

  const unsigned ktLds = (unsigned)(unsigned long long)(const void*)kt;

  const int jmax = (qbase + 127) >> 6;
  for (int j = 0; j <= jmax; ++j) {
    const int kb = j * 64;
    __syncthreads();  // previous tile fully consumed before restaging
    // ---- K tile via Tensor Data Mover (issued once, by wave 0) ----
    if (wave == 0) {
      tdm_load_tile_k64(Kp + bhoff + (size_t)kb * D, ktLds, D);
    }
    // ---- V tile staged transposed by all 256 threads ----
#pragma unroll
    for (int it = 0; it < 2; ++it) {
      int idx = (threadIdx.x + it * 256) * 8;  // covers 64x64 in 8-elt chunks
      int key = idx >> 6;
      int d = idx & 63;
      const u16* vsrc = Vp + bhoff + (size_t)(kb + key) * D + d;
      union { uint4 q; u16 s[8]; } vv;
      vv.q = *reinterpret_cast<const uint4*>(vsrc);
#pragma unroll
      for (int e = 0; e < 8; ++e) vt[(d + e) * LP + key] = vv.s[e];
    }
    if (wave == 0) __builtin_amdgcn_s_wait_tensorcnt(0);  // K tile landed
    __syncthreads();

    const bool active = (kb <= q0 + 15);  // causal: this wave needs this tile?
    if (active) {
      // ---- scores = Q(16xDK) * K^T(DKx64), already scaled via Q ----
      v8f s[4];
#pragma unroll
      for (int nt = 0; nt < 4; ++nt) s[nt] = vzero;
#pragma unroll
      for (int ks = 0; ks < 2; ++ks)
#pragma unroll
        for (int nt = 0; nt < 4; ++nt) {
          const u16* kp2 = &kt[(nt * 16 + l15) * LP + ks * 32 + hl * 8];
          v16bf bfr = load_frag(kp2, kp2 + 16);
          s[nt] = __builtin_amdgcn_wmma_f32_16x16x32_bf16(
              false, qf[ks], false, bfr, (short)0, s[nt], false, false);
        }
      // ---- causal mask (diagonal tiles only) ----
      if (kb + 63 > q0) {
#pragma unroll
        for (int nt = 0; nt < 4; ++nt)
#pragma unroll
          for (int r = 0; r < 8; ++r) {
            if (kb + nt * 16 + l15 > q0 + r + 8 * hl) s[nt][r] = -3.0e38f;
          }
      }
      // ---- online softmax; row r+8*hl lives in VGPR r across the half-wave ----
      u16* pw = &pt[wave * 16 * LP];
#pragma unroll
      for (int r = 0; r < 8; ++r) {
        float tm = s[0][r];
#pragma unroll
        for (int nt = 1; nt < 4; ++nt) tm = fmaxf(tm, s[nt][r]);
#pragma unroll
        for (int off = 8; off > 0; off >>= 1) tm = fmaxf(tm, __shfl_xor(tm, off, 16));
        float mnew = fmaxf(mrow[r], tm);
        float sf = __expf(mrow[r] - mnew);
        float rs = 0.0f;
#pragma unroll
        for (int nt = 0; nt < 4; ++nt) {
          float p = __expf(s[nt][r] - mnew);
          s[nt][r] = p;
          rs += p;
        }
#pragma unroll
        for (int off = 8; off > 0; off >>= 1) rs += __shfl_xor(rs, off, 16);
        lrow[r] = lrow[r] * sf + rs;
        mrow[r] = mnew;
#pragma unroll
        for (int nt = 0; nt < 4; ++nt) oacc[nt][r] *= sf;
      }
      // ---- spill P (bf16) to LDS in A-fragment-friendly row-major layout ----
#pragma unroll
      for (int nt = 0; nt < 4; ++nt)
#pragma unroll
        for (int r = 0; r < 8; ++r)
          pw[(r + 8 * hl) * LP + nt * 16 + l15] = f2bf(s[nt][r]);
    }
    __syncthreads();  // P visible (also keeps block barrier-uniform)
    if (active) {
      // ---- O += P(16x64) * V(64xDK) ----
      const u16* pw = &pt[wave * 16 * LP];
#pragma unroll
      for (int ks2 = 0; ks2 < 2; ++ks2) {
        const u16* pp = &pw[l15 * LP + ks2 * 32 + hl * 8];
        v16bf pa = load_frag(pp, pp + 16);
#pragma unroll
        for (int nt = 0; nt < 4; ++nt) {
          const u16* vp2 = &vt[(nt * 16 + l15) * LP + ks2 * 32 + hl * 8];
          v16bf vb2 = load_frag(vp2, vp2 + 16);
          oacc[nt] = __builtin_amdgcn_wmma_f32_16x16x32_bf16(
              false, pa, false, vb2, (short)0, oacc[nt], false, false);
        }
      }
    }
  }

  // ---- normalize and store O as bf16 in (B*S, D) row-major (head-merged) ----
#pragma unroll
  for (int nt = 0; nt < 4; ++nt)
#pragma unroll
    for (int r = 0; r < 8; ++r) {
      int q = q0 + r + 8 * hl;
      float v = oacc[nt][r] / lrow[r];
      Ob[bhoff + (size_t)q * D + nt * 16 + l15] = f2bf(v);
    }
}

// ---------------------------------------------------------------- launcher
extern "C" void kernel_launch(void* const* d_in, const int* in_sizes, int n_in,
                              void* d_out, int out_size, void* d_ws, size_t ws_size,
                              hipStream_t stream) {
  const float* x  = (const float*)d_in[0];
  const float* wq = (const float*)d_in[1];
  const float* wk = (const float*)d_in[2];
  const float* wv = (const float*)d_in[3];
  const float* wo = (const float*)d_in[4];
  float* out = (float*)d_out;

  constexpr int Bb = 2, S = 2048, D = 1024;
  constexpr int MS = Bb * S;  // 4096 rows

  char* ws = (char*)d_ws;
  u16* xb  = (u16*)(ws + (size_t)0);          // 8 MiB  x  bf16
  u16* wqb = (u16*)(ws + ((size_t)8  << 20)); // 2 MiB each weight
  u16* wkb = (u16*)(ws + ((size_t)10 << 20));
  u16* wvb = (u16*)(ws + ((size_t)12 << 20));
  u16* wob = (u16*)(ws + ((size_t)14 << 20));
  u16* Qb  = (u16*)(ws + ((size_t)16 << 20)); // 8 MiB each of Q,K,V,O
  u16* Kb  = (u16*)(ws + ((size_t)24 << 20));
  u16* Vb  = (u16*)(ws + ((size_t)32 << 20));
  u16* Ob  = (u16*)(ws + ((size_t)40 << 20)); // total 48 MiB

  cvt_f32_bf16<<<1024, 256, 0, stream>>>(x,  xb,  MS * D / 4);
  cvt_f32_bf16<<<512,  256, 0, stream>>>(wq, wqb, D * D / 4);
  cvt_f32_bf16<<<512,  256, 0, stream>>>(wk, wkb, D * D / 4);
  cvt_f32_bf16<<<512,  256, 0, stream>>>(wv, wvb, D * D / 4);
  cvt_f32_bf16<<<512,  256, 0, stream>>>(wo, wob, D * D / 4);

  dim3 gg(D / 128, MS / 128);  // (8, 32) = 256 blocks
  // Q projection absorbs the 1/sqrt(dk)=0.125 softmax scale.
  gemm_xwT<0><<<gg, 128, 0, stream>>>(xb, wqb, Qb, MS, D, D, 0.125f);
  gemm_xwT<0><<<gg, 128, 0, stream>>>(xb, wkb, Kb, MS, D, D, 1.0f);
  gemm_xwT<0><<<gg, 128, 0, stream>>>(xb, wvb, Vb, MS, D, D, 1.0f);

  flash_attn<<<dim3(Bb * 16, S / 128), 256, 0, stream>>>(Qb, Kb, Vb, Ob);

  gemm_xwT<1><<<gg, 128, 0, stream>>>(Ob, wob, out, MS, D, D, 1.0f);
}
